// SchnetConv_54176717472000
// MI455X (gfx1250) — matrix operand order repaired and checked by reading, not compiled, not run on hardware
//
#include <hip/hip_runtime.h>

// ---------------------------------------------------------------------------
// SchNet conv, fused for MI455X (gfx1250, wave32, WMMA).
// Memory-bound (~1 GB traffic -> ~45us @ 23.3 TB/s); matmuls on
// v_wmma_f32_16x16x32_bf16 with f32 accumulation. Single streaming pass over
// edges; nh (25.6MB) and the scatter target (25.6MB) live in the 192MB L2.
// d_out doubles as the segment-sum accumulator; d_ws holds N floats of counts.
// ---------------------------------------------------------------------------

#define DF   64     // in/out feature dim
#define RADF 128    // radial basis dim
#define LOG2C 0.69314718055994530942f

typedef __attribute__((ext_vector_type(16))) __bf16 v16bf;
typedef __attribute__((ext_vector_type(16))) short  v16s;
typedef __attribute__((ext_vector_type(8)))  short  v8s;
typedef __attribute__((ext_vector_type(8)))  float  v8f;
typedef __attribute__((ext_vector_type(16))) float  v16f;
typedef __attribute__((ext_vector_type(4)))  float  v4f;

__device__ __forceinline__ short f2bf(float f) {
  return __builtin_bit_cast(short, (__bf16)f);   // native v_cvt, RNE
}

// four 4-float loads -> one bf16 A-fragment half-row (packed converts)
__device__ __forceinline__ v16bf pack16(v4f x0, v4f x1, v4f x2, v4f x3) {
  v8f lo = __builtin_shufflevector(x0, x1, 0,1,2,3,4,5,6,7);
  v8f hi = __builtin_shufflevector(x2, x3, 0,1,2,3,4,5,6,7);
  v16f xf = __builtin_shufflevector(lo, hi, 0,1,2,3,4,5,6,7,8,9,10,11,12,13,14,15);
  return __builtin_convertvector(xf, v16bf);
}

__device__ __forceinline__ float ssp(float x) { // softplus(x) - log(2)
  if (x > 15.f) return x - LOG2C;
  return __logf(1.f + __expf(x)) - LOG2C;
}

// Stage a [K x 64] f32 weight matrix into LDS as bf16, pre-swizzled into
// WMMA B fragments (16x16x32 bf16): frag (nt,c) holds B[k=c*32..+31][n=nt*16..+15].
// Lane L of a frag holds halves[h] = B[c*32 + (L>>4)*8 + (h&7) + (h>>3)*16][ (L&15) + nt*16 ],
// stored contiguously so each lane reads its 16 halves as one 32B LDS access.
__device__ void stage_weights(const float* __restrict__ W, short* dst, int nchunks) {
  int total = 4 * nchunks * 512;
  for (int i = threadIdx.x; i < total; i += blockDim.x) {
    int f = i >> 9, r = i & 511;
    int nt = f / nchunks, c = f % nchunks;
    int lane = r >> 4, h = r & 15;
    int n = (lane & 15) + nt * 16;
    int k = c * 32 + (lane >> 4) * 8 + (h & 7) + (h >> 3) * 16;
    dst[i] = f2bf(W[k * DF + n]);
  }
}

__global__ void schnet_zero_kernel(float* __restrict__ sacc,
                                   float* __restrict__ cnt, int N) {
  size_t total  = (size_t)N * DF;
  size_t i      = (size_t)blockIdx.x * blockDim.x + threadIdx.x;
  size_t stride = (size_t)gridDim.x * blockDim.x;
  for (size_t j = i; j < total;     j += stride) sacc[j] = 0.f;
  for (size_t j = i; j < (size_t)N; j += stride) cnt[j]  = 0.f;
}

__global__ void schnet_count_kernel(const int* __restrict__ dst,
                                    float* __restrict__ cnt, int E) {
  int i = blockIdx.x * blockDim.x + threadIdx.x;
  if (i < E) atomicAdd(&cnt[dst[i]], 1.0f);
}

// ---------------------------------------------------------------------------
// Edge pipeline: per wave, one 16-edge tile.
//   f   = ssp(bf@W1+b1); f = ssp(f@W2+b2)      (WMMA bf16, f32 acc)
//   m   = nh[src] * (f * eh)                   (fused, C-layout)
//   sacc[dst] += m                             (global_atomic_add_f32)
// ---------------------------------------------------------------------------
__global__ __launch_bounds__(256) void schnet_edge_kernel(
    const float* __restrict__ bf, const float* __restrict__ eh,
    const float* __restrict__ nh, const int* __restrict__ src,
    const int* __restrict__ dst,
    const float* __restrict__ W1, const float* __restrict__ b1,
    const float* __restrict__ W2, const float* __restrict__ b2,
    float* __restrict__ sacc, int E)
{
  __shared__ __attribute__((aligned(16))) short ldsW1[16 * 512]; // 16KB
  __shared__ __attribute__((aligned(16))) short ldsW2[8 * 512];  // 8KB
  __shared__ __attribute__((aligned(16))) short bounce[8 * 16 * DF]; // 16KB

  stage_weights(W1, ldsW1, 4);
  stage_weights(W2, ldsW2, 2);
  __syncthreads();

  const int lane = threadIdx.x & 31;
  const int wave = threadIdx.x >> 5;
  const int half = lane >> 4;   // lanes 16-31 handle the second K/M half
  const int mrow = lane & 15;
  short* myB = bounce + wave * 16 * DF;

  const int nTiles = E >> 4;
  const int gw = blockIdx.x * (blockDim.x >> 5) + wave;
  const int nw = gridDim.x * (blockDim.x >> 5);

  for (int t = gw; t < nTiles; t += nw) {
    const int e0 = t << 4;
    const float* rowp = bf + (size_t)(e0 + mrow) * RADF + half * 8;

    // prefetch next tile's bf slice (global_prefetch_b8)
    int tn = t + nw;
    if (tn < nTiles) {
      const float* np = bf + (size_t)((tn << 4) + mrow) * RADF + half * 8;
      #pragma unroll
      for (int c = 0; c < 4; ++c) __builtin_prefetch(np + c * 32, 0, 1);
    }

    // ---- layer 1: [16x128]bf16 @ W1 -> [16x64] f32 ----
    v8f acc[4] = {};
    #pragma unroll
    for (int c = 0; c < 4; ++c) {
      // A-fragment (16x32 bf16): halves 0-7 = K c*32+half*8+j,
      //                          halves 8-15 = K c*32+16+half*8+j
      v4f x0 = *(const v4f*)(rowp + c * 32);
      v4f x1 = *(const v4f*)(rowp + c * 32 + 4);
      v4f x2 = *(const v4f*)(rowp + c * 32 + 16);
      v4f x3 = *(const v4f*)(rowp + c * 32 + 20);
      v16bf a = pack16(x0, x1, x2, x3);
      #pragma unroll
      for (int nt = 0; nt < 4; ++nt) {
        v16bf b = __builtin_bit_cast(v16bf,
            *(const v16s*)(ldsW1 + (nt * 4 + c) * 512 + lane * 16));
        acc[nt] = __builtin_amdgcn_wmma_f32_16x16x32_bf16(
            false, a, false, b, (short)0, acc[nt], false, false);
      }
    }

    // bias + ssp; park bf16 result in per-wave LDS scratch for re-layout
    #pragma unroll
    for (int nt = 0; nt < 4; ++nt) {
      float bias = b1[nt * 16 + mrow];
      #pragma unroll
      for (int r = 0; r < 8; ++r) {
        float v = ssp(acc[nt][r] + bias);
        myB[(half * 8 + r) * DF + nt * 16 + mrow] = f2bf(v);
      }
    }
    asm volatile("s_wait_dscnt 0" ::: "memory"); // same-wave LDS RAW fence

    // ---- layer 2: [16x64]bf16 @ W2 -> [16x64] f32 ----
    v8f acc2[4] = {};
    #pragma unroll
    for (int c = 0; c < 2; ++c) {
      const short* fp = myB + mrow * DF + c * 32 + half * 8;
      v8s lo = *(const v8s*)fp;
      v8s hi = *(const v8s*)(fp + 16);
      v16bf a = __builtin_bit_cast(v16bf, __builtin_shufflevector(
          lo, hi, 0,1,2,3,4,5,6,7,8,9,10,11,12,13,14,15));
      #pragma unroll
      for (int nt = 0; nt < 4; ++nt) {
        v16bf b = __builtin_bit_cast(v16bf,
            *(const v16s*)(ldsW2 + (nt * 2 + c) * 512 + lane * 16));
        acc2[nt] = __builtin_amdgcn_wmma_f32_16x16x32_bf16(
            false, a, false, b, (short)0, acc2[nt], false, false);
      }
    }

    // ---- gating + gather + scatter, directly from C-layout ----
    int4 s4a = *(const int4*)(src + e0 + half * 8);
    int4 s4b = *(const int4*)(src + e0 + half * 8 + 4);
    int4 d4a = *(const int4*)(dst + e0 + half * 8);
    int4 d4b = *(const int4*)(dst + e0 + half * 8 + 4);
    int srcs[8] = {s4a.x, s4a.y, s4a.z, s4a.w, s4b.x, s4b.y, s4b.z, s4b.w};
    int dsts[8] = {d4a.x, d4a.y, d4a.z, d4a.w, d4b.x, d4b.y, d4b.z, d4b.w};
    const float* ehp = eh + (size_t)e0 * DF;

    #pragma unroll
    for (int nt = 0; nt < 4; ++nt) {
      float bias = b2[nt * 16 + mrow];
      int feat = nt * 16 + mrow;
      #pragma unroll
      for (int r = 0; r < 8; ++r) {
        int row = half * 8 + r;                       // C-layout: M = r + 8*half
        float g = ssp(acc2[nt][r] + bias);
        float m = g * ehp[(size_t)row * DF + feat]
                    * nh[(size_t)srcs[r] * DF + feat];
        atomicAdd(&sacc[(size_t)dsts[r] * DF + feat], m);
      }
    }
  }
}

// ---------------------------------------------------------------------------
// Node MLP: agg = where(cnt>0, s/max(cnt,1), 0); out = ssp(ssp(agg@W3+b3)@W4+b4)
// In-place on `out` (each 16-node tile only touches its own rows).
// ---------------------------------------------------------------------------
__global__ __launch_bounds__(256) void schnet_node_kernel(
    const float* __restrict__ W3, const float* __restrict__ b3,
    const float* __restrict__ W4, const float* __restrict__ b4,
    const float* __restrict__ cnt, float* __restrict__ out, int N)
{
  __shared__ __attribute__((aligned(16))) short ldsW3[8 * 512];
  __shared__ __attribute__((aligned(16))) short ldsW4[8 * 512];
  __shared__ __attribute__((aligned(16))) short bounce[8 * 16 * DF];

  stage_weights(W3, ldsW3, 2);
  stage_weights(W4, ldsW4, 2);
  __syncthreads();

  const int lane = threadIdx.x & 31;
  const int wave = threadIdx.x >> 5;
  const int half = lane >> 4;
  const int mrow = lane & 15;
  short* myB = bounce + wave * 16 * DF;

  const int nTiles = N >> 4;
  const int gw = blockIdx.x * (blockDim.x >> 5) + wave;
  const int nw = gridDim.x * (blockDim.x >> 5);

  for (int t = gw; t < nTiles; t += nw) {
    const int n0 = t << 4;
    const int myrow = n0 + mrow;
    float c0 = cnt[myrow];
    float inv = (c0 > 0.f) ? (1.f / fmaxf(c0, 1.f)) : 0.f;
    const float* rowp = out + (size_t)myrow * DF + half * 8;

    v8f acc[4] = {};
    #pragma unroll
    for (int c = 0; c < 2; ++c) {
      v4f x0 = *(const v4f*)(rowp + c * 32) * inv;
      v4f x1 = *(const v4f*)(rowp + c * 32 + 4) * inv;
      v4f x2 = *(const v4f*)(rowp + c * 32 + 16) * inv;
      v4f x3 = *(const v4f*)(rowp + c * 32 + 20) * inv;
      v16bf a = pack16(x0, x1, x2, x3);
      #pragma unroll
      for (int nt = 0; nt < 4; ++nt) {
        v16bf b = __builtin_bit_cast(v16bf,
            *(const v16s*)(ldsW3 + (nt * 2 + c) * 512 + lane * 16));
        acc[nt] = __builtin_amdgcn_wmma_f32_16x16x32_bf16(
            false, a, false, b, (short)0, acc[nt], false, false);
      }
    }

    #pragma unroll
    for (int nt = 0; nt < 4; ++nt) {
      float bias = b3[nt * 16 + mrow];
      #pragma unroll
      for (int r = 0; r < 8; ++r) {
        float v = ssp(acc[nt][r] + bias);
        myB[(half * 8 + r) * DF + nt * 16 + mrow] = f2bf(v);
      }
    }
    asm volatile("s_wait_dscnt 0" ::: "memory");

    v8f acc2[4] = {};
    #pragma unroll
    for (int c = 0; c < 2; ++c) {
      const short* fp = myB + mrow * DF + c * 32 + half * 8;
      v8s lo = *(const v8s*)fp;
      v8s hi = *(const v8s*)(fp + 16);
      v16bf a = __builtin_bit_cast(v16bf, __builtin_shufflevector(
          lo, hi, 0,1,2,3,4,5,6,7,8,9,10,11,12,13,14,15));
      #pragma unroll
      for (int nt = 0; nt < 4; ++nt) {
        v16bf b = __builtin_bit_cast(v16bf,
            *(const v16s*)(ldsW4 + (nt * 2 + c) * 512 + lane * 16));
        acc2[nt] = __builtin_amdgcn_wmma_f32_16x16x32_bf16(
            false, a, false, b, (short)0, acc2[nt], false, false);
      }
    }

    #pragma unroll
    for (int nt = 0; nt < 4; ++nt) {
      float bias = b4[nt * 16 + mrow];
      int feat = nt * 16 + mrow;
      #pragma unroll
      for (int r = 0; r < 8; ++r) {
        int row = n0 + half * 8 + r;
        out[(size_t)row * DF + feat] = ssp(acc2[nt][r] + bias);
      }
    }
  }
}

extern "C" void kernel_launch(void* const* d_in, const int* in_sizes, int n_in,
                              void* d_out, int out_size, void* d_ws, size_t ws_size,
                              hipStream_t stream) {
  (void)n_in; (void)out_size; (void)ws_size;
  const float* bf  = (const float*)d_in[0];
  const float* eh  = (const float*)d_in[1];
  const float* nh  = (const float*)d_in[2];
  const int*   src = (const int*)d_in[3];
  const int*   dst = (const int*)d_in[4];
  const float* W1  = (const float*)d_in[5];
  const float* b1  = (const float*)d_in[6];
  const float* W2  = (const float*)d_in[7];
  const float* b2  = (const float*)d_in[8];
  const float* W3  = (const float*)d_in[9];
  const float* b3  = (const float*)d_in[10];
  const float* W4  = (const float*)d_in[11];
  const float* b4  = (const float*)d_in[12];

  const int E = in_sizes[3];          // number of edges
  const int N = in_sizes[2] / DF;     // number of nodes

  float* out = (float*)d_out;         // doubles as the f32 segment-sum buffer
  float* cnt = (float*)d_ws;          // N floats of scratch

  schnet_zero_kernel<<<2048, 256, 0, stream>>>(out, cnt, N);
  schnet_count_kernel<<<(E + 255) / 256, 256, 0, stream>>>(dst, cnt, E);
  schnet_edge_kernel<<<960, 256, 0, stream>>>(bf, eh, nh, src, dst,
                                              W1, b1, W2, b2, out, E);
  schnet_node_kernel<<<400, 256, 0, stream>>>(W3, b3, W4, b4, cnt, out, N);
}
